// AttentionCross_38139309588530
// MI455X (gfx1250) — matrix operand order repaired and checked by reading, not compiled, not run on hardware
//
#include <hip/hip_runtime.h>

#define DIM     1024
#define NHEADS  16
#define HDIM    64
#define SEQ_N   2048
#define SEQ_M   2048
#define BATCH   2
#define ATTN_SCALE 0.125f   // 64^-0.5

typedef __bf16 bf16_t;
typedef __attribute__((ext_vector_type(16))) __bf16 v16bf;
typedef __attribute__((ext_vector_type(8)))  __bf16 v8bf;
typedef __attribute__((ext_vector_type(8)))  float  v8f;

#if defined(__gfx1250__) && __has_builtin(__builtin_amdgcn_tensor_load_to_lds)
#define HAVE_TDM 1
#else
#define HAVE_TDM 0
#endif

// ---------------------------------------------------------------------------
// Tensor Data Mover: one 2D-tile DMA (global -> LDS), per CDNA5 ISA D# spec.
// Group0: count=1 | lds_addr[63:32] | global_addr[120:64] | type=2 [127:126]
// Group1: data_size=2B | tensor_dim0 | tensor_dim1 | tile_dim0/1 | dim0 stride
// ---------------------------------------------------------------------------
#if HAVE_TDM
typedef __attribute__((ext_vector_type(4))) unsigned int u32x4;
typedef __attribute__((ext_vector_type(8))) int          i32x8;
typedef __attribute__((ext_vector_type(4))) int          i32x4;

__device__ inline void tdm_load_2d(bf16_t* lds, const bf16_t* g,
                                   int rows, int rowlen, int gstride) {
  unsigned long long ga = (unsigned long long)(uintptr_t)g;
  u32x4 g0;
  g0[0] = 1u;                                        // count=1 (valid descriptor)
  g0[1] = (unsigned)(uintptr_t)lds;                  // lds_addr (low 32 bits)
  g0[2] = (unsigned)ga;                              // global_addr[31:0]
  g0[3] = ((unsigned)(ga >> 32) & 0x01FFFFFFu)       // global_addr[56:32]
          | 0x80000000u;                             // type=2 ("image")
  i32x8 g1 = {0, 0, 0, 0, 0, 0, 0, 0};
  g1[0] = (int)(1u << 16);                           // data_size=1 -> 2 bytes
  g1[1] = (int)(((unsigned)gstride & 0xFFFFu) << 16);// tensor_dim0[15:0]
  g1[2] = (int)(((unsigned)gstride >> 16) & 0xFFFFu);// tensor_dim0[31:16]
  g1[3] = (int)((1u << 14)                           // tensor_dim1 = 2^30 rows
                | (((unsigned)rowlen & 0xFFFFu) << 16)); // tile_dim0
  g1[4] = (int)((unsigned)rows & 0xFFFFu);           // tile_dim1
  g1[5] = (int)(unsigned)gstride;                    // tensor_dim0_stride[31:0]
  i32x4 gz = {0, 0, 0, 0};
#if __clang_major__ >= 23
  i32x8 gz8 = {0, 0, 0, 0, 0, 0, 0, 0};
  __builtin_amdgcn_tensor_load_to_lds(g0, g1, gz, gz, gz8, 0);
#else
  __builtin_amdgcn_tensor_load_to_lds(g0, g1, gz, gz, 0);
#endif
}
#endif

// Stage a [rows x rowlen] bf16 tile (contiguous rows in LDS) from global with
// row pitch gstride. TDM path: wave 0 issues one DMA. Fallback: b128 copies.
__device__ inline void tile_load(bf16_t* lds, const bf16_t* g,
                                 int rows, int rowlen, int gstride, int tid) {
#if HAVE_TDM
  if (tid < 32) tdm_load_2d(lds, g, rows, rowlen, gstride);
#else
  int cpr = rowlen >> 3;  // b128 chunks per row
  for (int c = tid; c < rows * cpr; c += (int)blockDim.x) {
    int r = c / cpr, o = (c - r * cpr) * 8;
    *(v8bf*)(lds + r * rowlen + o) = *(const v8bf*)(g + (size_t)r * gstride + o);
  }
#endif
}

__device__ inline void tile_wait(int tid) {
#if HAVE_TDM
  if (tid < 32) __builtin_amdgcn_s_wait_tensorcnt(0);
#endif
}

// ---------------------------------------------------------------------------
// WMMA fragment helpers (ISA VGPR layouts, CDNA5 ISA sec. 7.12.2)
// ---------------------------------------------------------------------------
__device__ inline v16bf frag_cat(v8bf lo, v8bf hi) {
  return __builtin_shufflevector(lo, hi, 0, 1, 2, 3, 4, 5, 6, 7,
                                 8, 9, 10, 11, 12, 13, 14, 15);
}

// A fragment (16x32): lanes 0-15 hold K 0..7 & 16..23, lanes 16-31 shifted by 8.
__device__ inline v16bf load_a_frag(const bf16_t* p, int row0, int ld, int k0) {
  int lane = threadIdx.x & 31;
  int r  = row0 + (lane & 15);
  int kb = k0 + ((lane >> 4) << 3);
  v8bf lo = *(const v8bf*)(p + r * ld + kb);
  v8bf hi = *(const v8bf*)(p + r * ld + kb + 16);
  return frag_cat(lo, hi);
}

// B fragment (32x16) from LDS stored transposed [n][k]: lanes 0-15 hold K 0..7 & 16..23, lanes 16-31 shifted by 8.
__device__ inline v16bf load_b_frag(const bf16_t* p, int n0, int ld, int k0) {
  int lane = threadIdx.x & 31;
  int n  = n0 + (lane & 15);
  int kb = k0 + ((lane >> 4) << 3);
  v8bf lo = *(const v8bf*)(p + n * ld + kb);
  v8bf hi = *(const v8bf*)(p + n * ld + kb + 16);
  return frag_cat(lo, hi);
}

__device__ inline v8f wmma_bf16(v16bf a, v16bf b, v8f c) {
  return __builtin_amdgcn_wmma_f32_16x16x32_bf16(false, a, false, b,
                                                 (short)0, c, false, false);
}

// ---------------------------------------------------------------------------
// Elementwise f32 -> bf16 (8 elems/thread, b128 in / b128 out)
// ---------------------------------------------------------------------------
__global__ __launch_bounds__(256) void convert_bf16_kernel(
    const float* __restrict__ in, bf16_t* __restrict__ out, int n) {
  int i = ((int)blockIdx.x * 256 + (int)threadIdx.x) * 8;
  if (i + 8 <= n) {
    float4 a = *(const float4*)(in + i);
    float4 b = *(const float4*)(in + i + 4);
    v8bf o = {(bf16_t)a.x, (bf16_t)a.y, (bf16_t)a.z, (bf16_t)a.w,
              (bf16_t)b.x, (bf16_t)b.y, (bf16_t)b.z, (bf16_t)b.w};
    *(v8bf*)(out + i) = o;
  }
}

// Transpose-convert: W[K x Ncols] f32 -> Wt[Ncols x K] bf16 (32x32 LDS tiles).
__global__ __launch_bounds__(256) void transpose_convert_kernel(
    const float* __restrict__ W, bf16_t* __restrict__ Wt, int K, int Ncols) {
  __shared__ float tile[32][33];
  int n0 = blockIdx.x * 32, k0 = blockIdx.y * 32;
  for (int j = threadIdx.y; j < 32; j += 8)
    tile[j][threadIdx.x] = W[(size_t)(k0 + j) * Ncols + n0 + threadIdx.x];
  __syncthreads();
  for (int j = threadIdx.y; j < 32; j += 8)
    Wt[(size_t)(n0 + j) * K + k0 + threadIdx.x] = (bf16_t)tile[threadIdx.x][j];
}

// ---------------------------------------------------------------------------
// GEMM: Out[Mrows x Ncols] = A[Mrows x K](bf16) * Wt[Ncols x K](bf16) (+bias).
// 256 threads / 8 waves, 128x128 tile, wave = 32x64, BK=32.
// TDM double-buffered LDS staging: wait tensorcnt -> barrier -> issue next DMA
// into the other buffer -> WMMA on current buffer.
// VT=true: scatter output into V^T layout [B, H, D, M] (per-lane rows are 8
// consecutive m values, so each lane's stores are contiguous).
// ---------------------------------------------------------------------------
template <typename TOUT, bool VT>
__global__ __launch_bounds__(256) void gemm_bf16_wmma_kernel(
    const bf16_t* __restrict__ A, const bf16_t* __restrict__ Wt,
    const float* __restrict__ bias, TOUT* __restrict__ Out,
    int Mrows, int Ncols, int K) {
  __shared__ __align__(16) bf16_t As[2][128][32];
  __shared__ __align__(16) bf16_t Bs[2][128][32];

  const int tid   = threadIdx.x;
  const int lane  = tid & 31;
  const int wave  = tid >> 5;
  const int waveM = wave >> 1;
  const int waveN = wave & 1;
  const int rowBase = blockIdx.y * 128;
  const int colBase = blockIdx.x * 128;

  const bf16_t* Atile = A + (size_t)rowBase * K;
  const bf16_t* Btile = Wt + (size_t)colBase * K;

  const v8f vzero = {0.f, 0.f, 0.f, 0.f, 0.f, 0.f, 0.f, 0.f};
  v8f acc[2][4];
#pragma unroll
  for (int mt = 0; mt < 2; ++mt)
#pragma unroll
    for (int nt = 0; nt < 4; ++nt) acc[mt][nt] = vzero;

  // Prologue: stage k-tile 0 into buffer 0.
  tile_load(&As[0][0][0], Atile, 128, 32, K, tid);
  tile_load(&Bs[0][0][0], Btile, 128, 32, K, tid);

  const int nk = K / 32;
  for (int kt = 0; kt < nk; ++kt) {
    tile_wait(tid);     // current buffer's DMA complete (wave 0)
    __syncthreads();    // visible to all; prev compute done before overwrite
    const int cur = kt & 1;
    if (kt + 1 < nk) {
      tile_load(&As[cur ^ 1][0][0], Atile + (kt + 1) * 32, 128, 32, K, tid);
      tile_load(&Bs[cur ^ 1][0][0], Btile + (kt + 1) * 32, 128, 32, K, tid);
    }
#pragma unroll
    for (int mt = 0; mt < 2; ++mt) {
      v16bf a = load_a_frag(&As[cur][0][0], waveM * 32 + mt * 16, 32, 0);
#pragma unroll
      for (int nt = 0; nt < 4; ++nt) {
        v16bf b = load_b_frag(&Bs[cur][0][0], waveN * 64 + nt * 16, 32, 0);
        acc[mt][nt] = wmma_bf16(a, b, acc[mt][nt]);
      }
    }
  }

#pragma unroll
  for (int mt = 0; mt < 2; ++mt)
#pragma unroll
    for (int nt = 0; nt < 4; ++nt) {
      int col = colBase + waveN * 64 + nt * 16 + (lane & 15);
#pragma unroll
      for (int r = 0; r < 8; ++r) {
        int row = rowBase + waveM * 32 + mt * 16 + r + ((lane >> 4) << 3);
        float v = acc[mt][nt][r];
        if (VT) {
          // row = b*SEQ_M + m ; col = h*HDIM + d  ->  VT[b][h][d][m]
          size_t idx = (((size_t)(row >> 11) * NHEADS + (col >> 6)) * HDIM +
                        (col & (HDIM - 1))) * SEQ_M + (row & (SEQ_M - 1));
          Out[idx] = (TOUT)(bf16_t)v;
        } else if (bias) {
          Out[(size_t)row * Ncols + col] = (TOUT)(v + bias[col]);
        } else {
          Out[(size_t)row * Ncols + col] = (TOUT)(bf16_t)v;
        }
      }
    }
}

// ---------------------------------------------------------------------------
// Flash attention: block = 128 threads (4 waves) = one (b, h, 64-row q tile);
// wave owns 16 q rows. Q, K and V^T tiles all via TDM, K/V double-buffered.
// Kb: [B*M, H*D] bf16.  VTb: [B, H, D, M] bf16.
// ---------------------------------------------------------------------------
__global__ __launch_bounds__(128) void attn_fwd_kernel(
    const bf16_t* __restrict__ Qb, const bf16_t* __restrict__ Kb,
    const bf16_t* __restrict__ VTb, bf16_t* __restrict__ Ob) {
  __shared__ __align__(16) bf16_t Qs[64][64];         // [q][d]
  __shared__ __align__(16) bf16_t Ks[2][64][64];      // [m][d] (B^T: n=m,k=d)
  __shared__ __align__(16) bf16_t Vt[2][64][64];      // [d][m] (B^T: n=d,k=m)
  __shared__ __align__(16) bf16_t Ps[4][16][64];      // per-wave P staging

  const int tid  = threadIdx.x;
  const int lane = tid & 31;
  const int wave = tid >> 5;
  const int bid  = blockIdx.x;
  const int qt = bid & 31;
  const int h  = (bid >> 5) & 15;
  const int b  = bid >> 9;
  const int qbase = qt * 64;

  const bf16_t* Ktile = Kb + ((size_t)b * SEQ_M) * DIM + h * HDIM;   // pitch DIM
  const bf16_t* VTtile = VTb + ((size_t)(b * NHEADS + h) * HDIM) * SEQ_M; // pitch SEQ_M

  // Prologue: Q tile + first K/V chunk into buffer 0 (all TDM).
  tile_load(&Qs[0][0], Qb + ((size_t)b * SEQ_N + qbase) * DIM + h * HDIM,
            64, 64, DIM, tid);
  tile_load(&Ks[0][0][0], Ktile, 64, 64, DIM, tid);
  tile_load(&Vt[0][0][0], VTtile, 64, 64, SEQ_M, tid);

  const v8f vzero = {0.f, 0.f, 0.f, 0.f, 0.f, 0.f, 0.f, 0.f};
  float mrow[8], lrow[8];
  v8f Oacc[4];
#pragma unroll
  for (int r = 0; r < 8; ++r) { mrow[r] = -1e30f; lrow[r] = 0.0f; }
#pragma unroll
  for (int n = 0; n < 4; ++n) Oacc[n] = vzero;

  const int nchunks = SEQ_M / 64;
  for (int ct = 0; ct < nchunks; ++ct) {
    tile_wait(tid);     // current chunk's DMAs complete (wave 0)
    __syncthreads();    // visible to all; prev compute done before overwrite
    const int cur = ct & 1;
    if (ct + 1 < nchunks) {
      tile_load(&Ks[cur ^ 1][0][0], Ktile + (size_t)(ct + 1) * 64 * DIM,
                64, 64, DIM, tid);
      tile_load(&Vt[cur ^ 1][0][0], VTtile + (ct + 1) * 64,
                64, 64, SEQ_M, tid);
    }

    // S = Q * K^T  (16 q x 64 m per wave; D=64 -> 2 WMMA per 16x16 tile)
    v8f s[4];
#pragma unroll
    for (int t = 0; t < 4; ++t) s[t] = vzero;
#pragma unroll
    for (int kk = 0; kk < 64; kk += 32) {
      v16bf a = load_a_frag(&Qs[0][0], wave * 16, 64, kk);
#pragma unroll
      for (int t = 0; t < 4; ++t) {
        v16bf bb = load_b_frag(&Ks[cur][0][0], t * 16, 64, kk);
        s[t] = wmma_bf16(a, bb, s[t]);
      }
    }

    // Online softmax (C layout: row = 8*(lane>=16)+r, col = lane&15).
    const int half = lane >> 4;
#pragma unroll
    for (int r = 0; r < 8; ++r) {
      float mx = -1e30f;
#pragma unroll
      for (int t = 0; t < 4; ++t) mx = fmaxf(mx, s[t][r]);
#pragma unroll
      for (int off = 1; off < 16; off <<= 1)
        mx = fmaxf(mx, __shfl_xor(mx, off, 32));
      float mnew  = fmaxf(mrow[r], mx * ATTN_SCALE);
      float alpha = __expf(mrow[r] - mnew);
      float sum = 0.0f;
      int prow = half * 8 + r;
#pragma unroll
      for (int t = 0; t < 4; ++t) {
        float p = __expf(s[t][r] * ATTN_SCALE - mnew);
        sum += p;
        Ps[wave][prow][t * 16 + (lane & 15)] = (bf16_t)p;
      }
#pragma unroll
      for (int off = 1; off < 16; off <<= 1)
        sum += __shfl_xor(sum, off, 32);
      lrow[r] = lrow[r] * alpha + sum;
      mrow[r] = mnew;
#pragma unroll
      for (int n = 0; n < 4; ++n) Oacc[n][r] *= alpha;
    }

    // O += P * V
#pragma unroll
    for (int kk = 0; kk < 64; kk += 32) {
      v16bf a = load_a_frag(&Ps[wave][0][0], 0, 64, kk);
#pragma unroll
      for (int n = 0; n < 4; ++n) {
        v16bf bb = load_b_frag(&Vt[cur][0][0], n * 16, 64, kk);
        Oacc[n] = wmma_bf16(a, bb, Oacc[n]);
      }
    }
  }

  const size_t ooff = ((size_t)b * SEQ_N + qbase + wave * 16) * DIM + h * HDIM;
#pragma unroll
  for (int n = 0; n < 4; ++n)
#pragma unroll
    for (int r = 0; r < 8; ++r) {
      int row = ((lane >> 4) << 3) + r;
      float v = Oacc[n][r] / lrow[r];
      Ob[ooff + (size_t)row * DIM + n * 16 + (lane & 15)] = (bf16_t)v;
    }
}

// ---------------------------------------------------------------------------
extern "C" void kernel_launch(void* const* d_in, const int* in_sizes, int n_in,
                              void* d_out, int out_size, void* d_ws, size_t ws_size,
                              hipStream_t stream) {
  const float* x1    = (const float*)d_in[0];
  const float* x2    = (const float*)d_in[1];
  const float* Wq    = (const float*)d_in[2];
  const float* Wkv   = (const float*)d_in[3];
  const float* Wproj = (const float*)d_in[4];
  const float* bproj = (const float*)d_in[5];
  float* out = (float*)d_out;

  const int BN = BATCH * SEQ_N;  // 4096
  const int BM = BATCH * SEQ_M;  // 4096

  // Workspace (bf16): x1b,x2b | transposed weights | Q | K | V^T | O (~56 MB)
  bf16_t* p = (bf16_t*)d_ws;
  bf16_t* x1b    = p; p += (size_t)BN * DIM;
  bf16_t* x2b    = p; p += (size_t)BM * DIM;
  bf16_t* Wqt    = p; p += (size_t)DIM * DIM;
  bf16_t* Wkvt   = p; p += (size_t)2 * DIM * DIM;
  bf16_t* Wprojt = p; p += (size_t)DIM * DIM;
  bf16_t* Qb     = p; p += (size_t)BN * DIM;
  bf16_t* Kb     = p; p += (size_t)BM * DIM;
  bf16_t* VTb    = p; p += (size_t)BM * DIM;
  bf16_t* Ob     = p;

  // One-time conversions (HBM-bound, ~50 MB total traffic)
  convert_bf16_kernel<<<dim3((BN * DIM) / 2048), dim3(256), 0, stream>>>(
      x1, x1b, BN * DIM);
  convert_bf16_kernel<<<dim3((BM * DIM) / 2048), dim3(256), 0, stream>>>(
      x2, x2b, BM * DIM);
  transpose_convert_kernel<<<dim3(DIM / 32, DIM / 32), dim3(32, 8), 0, stream>>>(
      Wq, Wqt, DIM, DIM);
  transpose_convert_kernel<<<dim3(2 * DIM / 32, DIM / 32), dim3(32, 8), 0, stream>>>(
      Wkv, Wkvt, DIM, 2 * DIM);
  transpose_convert_kernel<<<dim3(DIM / 32, DIM / 32), dim3(32, 8), 0, stream>>>(
      Wproj, Wprojt, DIM, DIM);

  // Q = x1 @ Wq (bf16); K = x2 @ Wkv[:, :1024]; V^T from x2 @ Wkv[:, 1024:]
  gemm_bf16_wmma_kernel<bf16_t, false>
      <<<dim3(DIM / 128, BN / 128), dim3(256), 0, stream>>>(
          x1b, Wqt, nullptr, Qb, BN, DIM, DIM);
  gemm_bf16_wmma_kernel<bf16_t, false>
      <<<dim3(DIM / 128, BM / 128), dim3(256), 0, stream>>>(
          x2b, Wkvt, nullptr, Kb, BM, DIM, DIM);
  gemm_bf16_wmma_kernel<bf16_t, true>
      <<<dim3(DIM / 128, BM / 128), dim3(256), 0, stream>>>(
          x2b, Wkvt + (size_t)DIM * DIM, nullptr, VTb, BM, DIM, DIM);

  // Flash attention (Q/K/V^T tiles all via TDM, double-buffered)
  attn_fwd_kernel<<<dim3(BATCH * NHEADS * (SEQ_N / 64)), dim3(128), 0, stream>>>(
      Qb, Kb, VTb, Ob);

  // out = O @ Wproj + bproj (f32)
  gemm_bf16_wmma_kernel<float, false>
      <<<dim3(DIM / 128, BN / 128), dim3(256), 0, stream>>>(
          Ob, Wprojt, bproj, out, BN, DIM, DIM);
}
